// stackedIndRNN_encoder_20435454395119
// MI455X (gfx1250) — matrix operand order repaired and checked by reading, not compiled, not run on hardware
//
#include <hip/hip_runtime.h>

// ---------------------------------------------------------------------------
// Stacked IndRNN encoder for MI455X (gfx1250, wave32).
//   - GEMMs in bf16 via V_WMMA_F32_16X16X32_BF16 (fp32 accumulate)
//   - Tile staging via GLOBAL_LOAD_ASYNC_TO_LDS_B128 (ASYNCcnt) with
//     LDS double buffering; BK=64 stages (16 WMMAs per barrier pair)
//   - Steady-state loop peeled: no conditionals in the hot loop
//   - BN stats via block-parallel f32 atomics, BN-apply fused into IndRNN scan
//   - Weights pre-transposed to [N,K] bf16 so B-tiles load coalesced and each
//     lane's fragment is contiguous in LDS (contiguous k order, same as the A fragments)
// ---------------------------------------------------------------------------

typedef __attribute__((ext_vector_type(16))) __bf16 v16bf;
typedef __attribute__((ext_vector_type(8)))  __bf16 v8bf;
typedef __attribute__((ext_vector_type(8)))  float  v8f;

#define LNUM 6
#define TT   256
#define BB   64
#define DIN  128
#define HH   1024
#define CC   128
#define MM   (TT * BB)          // 16384 rows
#define EPSV 1e-5f

// GEMM tiling
#define BM 128
#define BN 128
#define BK 64                   // two 16x16x32 K-steps per LDS stage
#define LDT 80                  // padded LDS stride (bf16 elems); 160B rows

static __device__ __forceinline__ unsigned short f2bf(float f) {
    unsigned int u = __float_as_uint(f);
    // round-to-nearest-even
    unsigned int r = u + 0x7FFFu + ((u >> 16) & 1u);
    return (unsigned short)(r >> 16);
}

// ---------------------------------------------------------------------------
// fp32 -> bf16 elementwise
// ---------------------------------------------------------------------------
__global__ __launch_bounds__(256) void k_f32_to_bf16(const float* __restrict__ in,
                                                     unsigned short* __restrict__ out,
                                                     int n) {
    int i = blockIdx.x * 256 + threadIdx.x;
    if (i < n) out[i] = f2bf(in[i]);
}

// fp32 W[K,N] -> bf16 Wt[N,K]   (thread index: k fastest -> coalesced writes)
__global__ __launch_bounds__(256) void k_conv_transpose(const float* __restrict__ W,
                                                        unsigned short* __restrict__ Wt,
                                                        int K, int N) {
    int i = blockIdx.x * 256 + threadIdx.x;
    if (i >= K * N) return;
    int n = i / K;
    int k = i - n * K;
    Wt[i] = f2bf(W[(size_t)k * N + n]);
}

// ---------------------------------------------------------------------------
// bf16 GEMM: C[M,N] = A[M,K] * Bt[N,K]^T + bias[N]   (fp32 out)
// 256 threads = 8 waves; wave grid 4(M) x 2(N); wave tile 32x64.
// Double-buffered LDS fed by async global->LDS DMA; 16 WMMAs per stage.
// ---------------------------------------------------------------------------
__global__ __launch_bounds__(256) void k_gemm_bf16(const unsigned short* __restrict__ A,
                                                   const unsigned short* __restrict__ Bt,
                                                   const float* __restrict__ bias,
                                                   float* __restrict__ C,
                                                   int Mdim, int Ndim, int Kdim) {
    __shared__ __align__(32) unsigned short sA[2][BM * LDT];
    __shared__ __align__(32) unsigned short sB[2][BN * LDT];

    const int tid   = threadIdx.x;
    const int wave  = tid >> 5;
    const int lane  = tid & 31;
    const int l16   = lane & 15;
    const int hi    = lane >> 4;          // half-wave select
    const int waveM = wave & 3;           // 0..3 -> 32-row slab
    const int waveN = wave >> 2;          // 0..1 -> 64-col slab
    const int blockM = blockIdx.y * BM;
    const int blockN = blockIdx.x * BN;

    v8f acc[2][4];
#pragma unroll
    for (int i = 0; i < 2; ++i)
#pragma unroll
        for (int j = 0; j < 4; ++j) acc[i][j] = (v8f){};

    // issue 8 async global->LDS b128 loads for one double-buffer stage
    // (1024 16B chunks per 128x64 tile, 4 per thread per matrix)
    auto issue_tile = [&](int buf, int k0) {
#pragma unroll
        for (int i = 0; i < 4; ++i) {
            int chunk = tid + i * 256;        // 0..1023
            int row   = chunk >> 3;           // 0..127
            int kc    = (chunk & 7) * 8;      // 0..56
            const unsigned short* ga = A + (size_t)(blockM + row) * Kdim + k0 + kc;
            unsigned la = (unsigned)(size_t)(const void*)&sA[buf][row * LDT + kc];
            asm volatile("global_load_async_to_lds_b128 %0, %1, off"
                         :: "v"(la), "v"(ga) : "memory");
            const unsigned short* gb = Bt + (size_t)(blockN + row) * Kdim + k0 + kc;
            unsigned lb = (unsigned)(size_t)(const void*)&sB[buf][row * LDT + kc];
            asm volatile("global_load_async_to_lds_b128 %0, %1, off"
                         :: "v"(lb), "v"(gb) : "memory");
        }
    };

    // 16 WMMAs on one resident stage (two 16x16x32 K-steps)
    auto compute_stage = [&](int cur) {
#pragma unroll
        for (int ks = 0; ks < 2; ++ks) {
            const int kb = ks * 32;
            // A frags: lane<16 -> K0..15 contiguous; lane>=16 -> K16..31
            v16bf afrag[2];
#pragma unroll
            for (int mt = 0; mt < 2; ++mt) {
                const unsigned short* ab =
                    &sA[cur][(waveM * 32 + mt * 16 + l16) * LDT + kb];
                v8bf lo = *(const v8bf*)(const void*)(ab + hi * 16);
                v8bf hh = *(const v8bf*)(const void*)(ab + hi * 16 + 8);
                afrag[mt] = __builtin_shufflevector(lo, hh, 0, 1, 2, 3, 4, 5, 6, 7,
                                                            8, 9, 10, 11, 12, 13, 14, 15);
            }
            // B frags: lane<16 -> K0..15 contiguous; lane>=16 -> K16..31
            v16bf bfrag[4];
#pragma unroll
            for (int nt = 0; nt < 4; ++nt) {
                const unsigned short* bb =
                    &sB[cur][(waveN * 64 + nt * 16 + l16) * LDT + kb + hi * 16];
                bfrag[nt] = *(const v16bf*)(const void*)bb;
            }
#pragma unroll
            for (int mt = 0; mt < 2; ++mt)
#pragma unroll
                for (int nt = 0; nt < 4; ++nt)
                    acc[mt][nt] = __builtin_amdgcn_wmma_f32_16x16x32_bf16(
                        false, afrag[mt], false, bfrag[nt],
                        (short)0, acc[mt][nt], false, false);
        }
    };

    const int iters = Kdim / BK;
    issue_tile(0, 0);

    int it = 0;
    for (; it < iters - 1; ++it) {          // steady state: no conditionals
        const int cur = it & 1;
        issue_tile(cur ^ 1, (it + 1) * BK);
        // all but the 8 just-issued are done -> stage `it` has landed
        asm volatile("s_wait_asynccnt 0x8" ::: "memory");
        __syncthreads();                    // all waves' stage-`it` data visible
        compute_stage(cur);
        __syncthreads();                    // done reading before overwrite
    }
    // peeled last stage
    asm volatile("s_wait_asynccnt 0x0" ::: "memory");
    __syncthreads();
    compute_stage(it & 1);

    // epilogue: C/D layout — VGPR v: lanes 0-15 row=v, lanes 16-31 row=v+8
#pragma unroll
    for (int mt = 0; mt < 2; ++mt) {
#pragma unroll
        for (int nt = 0; nt < 4; ++nt) {
            int col   = blockN + waveN * 64 + nt * 16 + l16;
            int rbase = blockM + waveM * 32 + mt * 16 + hi * 8;
            float bv = bias[col];
#pragma unroll
            for (int v = 0; v < 8; ++v)
                C[(size_t)(rbase + v) * Ndim + col] = acc[mt][nt][v] + bv;
        }
    }
}

// ---------------------------------------------------------------------------
// BN statistics: per-channel sum / sumsq over 16384 rows via f32 atomics
// grid: (H/1024, M/BNROWS)   block: 256, each thread covers 4 channels (f4 ld)
// ---------------------------------------------------------------------------
#define BNROWS 64
__global__ __launch_bounds__(256) void k_bn_stats(const float* __restrict__ X,
                                                  float* __restrict__ sum,
                                                  float* __restrict__ sumsq) {
    int c4 = (blockIdx.x * 256 + threadIdx.x) * 4;
    int r0 = blockIdx.y * BNROWS;
    float s0 = 0.f, s1 = 0.f, s2 = 0.f, s3 = 0.f;
    float q0 = 0.f, q1 = 0.f, q2 = 0.f, q3 = 0.f;
    const float4* p = (const float4*)(X + (size_t)r0 * HH + c4);
    for (int r = 0; r < BNROWS; ++r) {
        float4 x = *p;
        s0 += x.x; q0 = fmaf(x.x, x.x, q0);
        s1 += x.y; q1 = fmaf(x.y, x.y, q1);
        s2 += x.z; q2 = fmaf(x.z, x.z, q2);
        s3 += x.w; q3 = fmaf(x.w, x.w, q3);
        p += HH / 4;
    }
    atomicAdd(&sum[c4 + 0], s0);
    atomicAdd(&sum[c4 + 1], s1);
    atomicAdd(&sum[c4 + 2], s2);
    atomicAdd(&sum[c4 + 3], s3);
    atomicAdd(&sumsq[c4 + 0], q0);
    atomicAdd(&sumsq[c4 + 1], q1);
    atomicAdd(&sumsq[c4 + 2], q2);
    atomicAdd(&sumsq[c4 + 3], q3);
}

__global__ __launch_bounds__(256) void k_bn_finalize(const float* __restrict__ sum,
                                                     const float* __restrict__ sumsq,
                                                     const float* __restrict__ gamma,
                                                     const float* __restrict__ beta,
                                                     float* __restrict__ scale,
                                                     float* __restrict__ shift) {
    int c = blockIdx.x * 256 + threadIdx.x;
    if (c >= HH) return;
    const float invN = 1.0f / (float)MM;
    float m  = sum[c] * invN;
    float v  = fmaf(sumsq[c], invN, -m * m);   // population variance
    float is = rsqrtf(v + EPSV);
    float sc = gamma[c] * is;
    scale[c] = sc;
    shift[c] = fmaf(-m, sc, beta[c]);
}

// ---------------------------------------------------------------------------
// Fused BN-apply + IndRNN scan. One thread per 2 adjacent (b,h) channels:
// float2 reads of X, packed 2xbf16 u32 writes of the activation.
// ---------------------------------------------------------------------------
__global__ __launch_bounds__(256) void k_indrnn(const float* __restrict__ X,
                                                const float* __restrict__ scale,
                                                const float* __restrict__ shift,
                                                const float* __restrict__ u,
                                                unsigned short* __restrict__ actOut,
                                                float* __restrict__ lastOut) {
    int idx  = blockIdx.x * 256 + threadIdx.x;   // pair index
    int pair = idx * 2;                          // b*H + h (h even)
    int h    = pair & (HH - 1);
    float sc0 = scale[h],  sc1 = scale[h + 1];
    float sh0 = shift[h],  sh1 = shift[h + 1];
    float uu0 = u[h],      uu1 = u[h + 1];
    float st0 = 0.f, st1 = 0.f;
    const size_t stride2 = (size_t)BB * HH / 2;  // in float2 / u32 units
    const float2* xp = (const float2*)(X + pair);
    unsigned int* op = (unsigned int*)(actOut + pair);
    for (int t = 0; t < TT; ++t) {
        float2 x = *xp;
        float v0 = fmaf(uu0, st0, fmaf(sc0, x.x, sh0));
        float v1 = fmaf(uu1, st1, fmaf(sc1, x.y, sh1));
        st0 = v0 > 0.f ? v0 : 0.f;
        st1 = v1 > 0.f ? v1 : 0.f;
        *op = (unsigned int)f2bf(st0) | ((unsigned int)f2bf(st1) << 16);
        xp += stride2;
        op += stride2;
    }
    lastOut[pair]     = st0;
    lastOut[pair + 1] = st1;
}

// ---------------------------------------------------------------------------
// Classifier: out[B,C] = last[B,H] * Wc[H,C] + bc   (16.8 MFLOP, negligible)
// ---------------------------------------------------------------------------
__global__ __launch_bounds__(256) void k_classifier(const float* __restrict__ last,
                                                    const float* __restrict__ Wc,
                                                    const float* __restrict__ bc,
                                                    float* __restrict__ out) {
    int idx = blockIdx.x * 256 + threadIdx.x;
    if (idx >= BB * CC) return;
    int b = idx / CC;
    int c = idx - b * CC;
    float s = bc[c];
    const float* lp = last + (size_t)b * HH;
    for (int k = 0; k < HH; ++k)
        s = fmaf(lp[k], Wc[(size_t)k * CC + c], s);
    out[idx] = s;
}

// ---------------------------------------------------------------------------
extern "C" void kernel_launch(void* const* d_in, const int* in_sizes, int n_in,
                              void* d_out, int out_size, void* d_ws, size_t ws_size,
                              hipStream_t stream) {
    (void)in_sizes; (void)n_in; (void)out_size; (void)ws_size;

    const float* inp   = (const float*)d_in[0];   // [T,B,DIN]
    const float* W0    = (const float*)d_in[1];   // [DIN,H]
    const float* b0    = (const float*)d_in[2];   // [H]
    const float* Ws    = (const float*)d_in[3];   // [L-1,H,H]
    const float* bs    = (const float*)d_in[4];   // [L-1,H]
    const float* u     = (const float*)d_in[5];   // [L,H]
    const float* gamma = (const float*)d_in[6];   // [L,H]
    const float* beta  = (const float*)d_in[7];   // [L,H]
    const float* Wc    = (const float*)d_in[8];   // [H,C]
    const float* bc    = (const float*)d_in[9];   // [C]
    float* out = (float*)d_out;                   // [B,C]

    char* ws = (char*)d_ws;
    size_t off = 0;
    unsigned short* act   = (unsigned short*)(ws + off); off += (size_t)MM * HH * 2;    // 32 MB
    float*          X     = (float*)(ws + off);          off += (size_t)MM * HH * 4;    // 64 MB
    unsigned short* inbf  = (unsigned short*)(ws + off); off += (size_t)MM * DIN * 2;   //  4 MB
    unsigned short* W0t   = (unsigned short*)(ws + off); off += (size_t)HH * DIN * 2;
    unsigned short* Wst   = (unsigned short*)(ws + off); off += (size_t)(LNUM - 1) * HH * HH * 2;
    float*          sum   = (float*)(ws + off);          off += HH * 4;
    float*          sumsq = (float*)(ws + off);          off += HH * 4;
    float*          scale = (float*)(ws + off);          off += HH * 4;
    float*          shift = (float*)(ws + off);          off += HH * 4;
    float*          last  = (float*)(ws + off);          off += (size_t)BB * HH * 4;

    // --- per-launch conversions (deterministic, done every call) ---
    {
        int n = MM * DIN;
        k_f32_to_bf16<<<(n + 255) / 256, 256, 0, stream>>>(inp, inbf, n);
    }
    {
        int n = DIN * HH;  // Wt[N=H][K=DIN]
        k_conv_transpose<<<(n + 255) / 256, 256, 0, stream>>>(W0, W0t, DIN, HH);
    }
    for (int l = 1; l < LNUM; ++l) {
        int n = HH * HH;
        k_conv_transpose<<<(n + 255) / 256, 256, 0, stream>>>(
            Ws + (size_t)(l - 1) * HH * HH, Wst + (size_t)(l - 1) * HH * HH, HH, HH);
    }

    dim3 gemmGrid(HH / BN, MM / BM);         // (8, 128)
    dim3 bnGrid(HH / 1024, MM / BNROWS);     // (1, 256)

    for (int l = 0; l < LNUM; ++l) {
        const unsigned short* A  = (l == 0) ? inbf : act;
        const unsigned short* Bt = (l == 0) ? W0t : (Wst + (size_t)(l - 1) * HH * HH);
        const float* bias        = (l == 0) ? b0 : (bs + (size_t)(l - 1) * HH);
        int K                    = (l == 0) ? DIN : HH;

        k_gemm_bf16<<<gemmGrid, 256, 0, stream>>>(A, Bt, bias, X, MM, HH, K);

        hipMemsetAsync(sum, 0, 2 * HH * sizeof(float), stream);  // sum + sumsq contiguous
        k_bn_stats<<<bnGrid, 256, 0, stream>>>(X, sum, sumsq);
        k_bn_finalize<<<HH / 256, 256, 0, stream>>>(sum, sumsq,
                                                    gamma + (size_t)l * HH,
                                                    beta + (size_t)l * HH,
                                                    scale, shift);

        k_indrnn<<<(BB * HH / 2) / 256, 256, 0, stream>>>(X, scale, shift,
                                                          u + (size_t)l * HH, act, last);
    }

    k_classifier<<<(BB * CC + 255) / 256, 256, 0, stream>>>(last, Wc, bc, out);
}
